// CausalSelfAttention_56942676411138
// MI455X (gfx1250) — matrix operand order repaired and checked by reading, not compiled, not run on hardware
//
#include <hip/hip_runtime.h>
#include <hip/hip_bf16.h>

typedef unsigned short u16t;
typedef __bf16 bf16_t;
typedef bf16_t v16bf __attribute__((ext_vector_type(16)));
typedef float  v8f   __attribute__((ext_vector_type(8)));

union Frag16 { v16bf v; uint4 q[2]; };

#define BT 8192   // B*T
#define TT 2048
#define CC 768
#define HH 12
#define DD 64

__device__ __forceinline__ u16t f2bf(float f) {
  unsigned u = __float_as_uint(f);
  u += 0x7FFFu + ((u >> 16) & 1u);          // round-to-nearest-even
  return (u16t)(u >> 16);
}
__device__ __forceinline__ float bf2f(u16t h) {
  return __uint_as_float(((unsigned)h) << 16);
}
__device__ __forceinline__ v8f zero8() {
  v8f z = {0.f,0.f,0.f,0.f,0.f,0.f,0.f,0.f};
  return z;
}

// ---------------------------------------------------------------------------
// CDNA5 async global -> LDS copy (16B per lane), tracked by ASYNCcnt.
// LDS destination offset comes from a generic -> addrspace(3) cast.
// ---------------------------------------------------------------------------
__device__ __forceinline__ void async_ld_b128(void* lds_ptr, const void* gptr) {
  unsigned l = (unsigned)(unsigned long long)(__attribute__((address_space(3))) char*)lds_ptr;
  asm volatile("global_load_async_to_lds_b128 %0, %1, off"
               :: "v"(l), "v"(gptr) : "memory");
}
__device__ __forceinline__ void wait_async0() {
  asm volatile("s_wait_asynccnt 0" ::: "memory");
}

// ---------------------------------------------------------------------------
// Kernel: fp32 -> bf16 cast
// ---------------------------------------------------------------------------
__global__ void cast_f32_bf16(const float* __restrict__ in, u16t* __restrict__ out, int n) {
  int i = blockIdx.x * blockDim.x + threadIdx.x;
  if (i < n) out[i] = f2bf(in[i]);
}

// ---------------------------------------------------------------------------
// Kernel: QKV GEMM  qkv[m,n] = sum_k xbf[m,k] * wqkv[n,k]
// Block tile 128x128, 8 waves (2x4), wave tile 64x32, K-step 32.
// Double-buffered async global->LDS pipeline; epilogue scatters into
// q/k/v buffers laid out (B,H,T,D) bf16.
// ---------------------------------------------------------------------------
__global__ void __launch_bounds__(256) gemm_qkv(const u16t* __restrict__ A,
                                                const u16t* __restrict__ W,
                                                u16t* __restrict__ qb,
                                                u16t* __restrict__ kb,
                                                u16t* __restrict__ vb) {
  const int K = CC;
  __shared__ alignas(16) u16t At[2][128 * 32];
  __shared__ alignas(16) u16t Bt[2][128 * 32];
  const int tid  = threadIdx.x;
  const int lane = tid & 31;
  const int wv   = tid >> 5;         // 0..7
  const int mw   = wv >> 2;          // 0..1
  const int nw   = wv & 3;           // 0..3
  const int g    = lane >> 4;        // half-wave group
  const int lr   = lane & 15;
  const int bm   = blockIdx.y * 128;
  const int bn   = blockIdx.x * 128;

  v8f acc[4][2];
#pragma unroll
  for (int i = 0; i < 4; ++i)
#pragma unroll
    for (int j = 0; j < 2; ++j) acc[i][j] = zero8();

  auto issue_tile = [&](int kt, int buf) {
#pragma unroll
    for (int c = 0; c < 2; ++c) {
      int ch  = tid + c * 256;                 // 0..511
      int row = ch >> 2;
      int col = (ch & 3) * 8;
      async_ld_b128(&At[buf][row * 32 + col], &A[(long)(bm + row) * K + kt + col]);
      async_ld_b128(&Bt[buf][row * 32 + col], &W[(long)(bn + row) * K + kt + col]);
    }
  };

  issue_tile(0, 0);
  int cur = 0;
  for (int kt = 0; kt < K; kt += 32) {
    wait_async0();                             // own tile DMA done
    __syncthreads();                           // everyone's tile done, prev buffer free
    if (kt + 32 < K) issue_tile(kt + 32, cur ^ 1);
    if (kt + 64 < K) {                         // warm L2 for the tile after next
      __builtin_prefetch(&A[(long)(bm + (tid >> 1)) * K + kt + 64], 0, 1);
      __builtin_prefetch(&W[(long)(bn + (tid >> 1)) * K + kt + 64], 0, 1);
    }

    const u16t* Ab = At[cur];
    const u16t* Bb = Bt[cur];
    Frag16 af[4], bfr[2];
#pragma unroll
    for (int mf = 0; mf < 4; ++mf) {
      const u16t* p = &Ab[(mw * 64 + mf * 16 + lr) * 32];
      af[mf].q[0] = *(const uint4*)&p[g * 8];
      af[mf].q[1] = *(const uint4*)&p[16 + g * 8];
    }
#pragma unroll
    for (int nf = 0; nf < 2; ++nf) {
      const u16t* p = &Bb[(nw * 32 + nf * 16 + lr) * 32];
      bfr[nf].q[0] = *(const uint4*)&p[g * 8];
      bfr[nf].q[1] = *(const uint4*)&p[16 + g * 8];
    }
#pragma unroll
    for (int mf = 0; mf < 4; ++mf)
#pragma unroll
      for (int nf = 0; nf < 2; ++nf)
        acc[mf][nf] = __builtin_amdgcn_wmma_f32_16x16x32_bf16(
            false, af[mf].v, false, bfr[nf].v, (short)0, acc[mf][nf], false, false);
    cur ^= 1;
  }

  // scatter: n -> which (q/k/v), head h, dim d ; m -> (b, t)
#pragma unroll
  for (int mf = 0; mf < 4; ++mf) {
#pragma unroll
    for (int nf = 0; nf < 2; ++nf) {
#pragma unroll
      for (int r = 0; r < 8; ++r) {
        int gm = bm + mw * 64 + mf * 16 + r + 8 * g;
        int gn = bn + nw * 32 + nf * 16 + lr;
        int which = gn / CC;
        int rem   = gn - which * CC;
        int h = rem >> 6;
        int d = rem & 63;
        int b = gm >> 11;
        int t = gm & (TT - 1);
        u16t val = f2bf(acc[mf][nf][r]);
        long idx = ((long)(b * HH + h) * TT + t) * DD + d;
        u16t* dst = (which == 0) ? qb : (which == 1 ? kb : vb);
        dst[idx] = val;
      }
    }
  }
}

// ---------------------------------------------------------------------------
// Kernel: RoPE on q and k (in place, bf16 storage, fp32 math)
// ---------------------------------------------------------------------------
__global__ void rope_kernel(u16t* __restrict__ qb, u16t* __restrict__ kb,
                            const float* __restrict__ sn, const float* __restrict__ cs) {
  int i = blockIdx.x * blockDim.x + threadIdx.x;   // over B*H*T*(D/2)
  if (i >= 4 * HH * TT * 32) return;
  int p = i & 31;
  int t = (i >> 5) & (TT - 1);
  long base = (long)(i >> 5) * DD + 2 * p;         // (bh*T + t)*D + 2p
  float s = sn[t * 32 + p], c = cs[t * 32 + p];
  float q1 = bf2f(qb[base]), q2 = bf2f(qb[base + 1]);
  qb[base]     = f2bf(q1 * c - q2 * s);
  qb[base + 1] = f2bf(q1 * s + q2 * c);
  float k1 = bf2f(kb[base]), k2 = bf2f(kb[base + 1]);
  kb[base]     = f2bf(k1 * c - k2 * s);
  kb[base + 1] = f2bf(k1 * s + k2 * c);
}

// ---------------------------------------------------------------------------
// Kernel: flash attention. Block = 4 waves, one (b,h) and a 64-row q tile.
// Each wave owns 16 q rows; KV streamed in causal 32-wide chunks.
// Q tile and K chunks use async global->LDS copies.
// ---------------------------------------------------------------------------
__global__ void __launch_bounds__(128) flash_attn(const u16t* __restrict__ qb,
                                                  const u16t* __restrict__ kb,
                                                  const u16t* __restrict__ vb,
                                                  u16t* __restrict__ aout) {
  __shared__ alignas(16) u16t Qs[64 * 64];       // q tile, row-major [t][d]
  __shared__ alignas(16) u16t Ks[32 * 64];       // kv chunk, [kv][d] (== B-frag layout)
  __shared__ alignas(16) u16t Vt[64 * 40];       // V transposed, [d][kv], stride 40
  __shared__ alignas(16) u16t Ps[4 * 16 * 40];   // per-wave P staging, stride 40

  const int tid  = threadIdx.x;
  const int lane = tid & 31;
  const int wv   = tid >> 5;                     // 0..3
  const int g    = lane >> 4;
  const int lr   = lane & 15;
  const int q0   = blockIdx.x * 64;
  const int bh   = blockIdx.y;                   // b*H + h
  const int b    = bh / HH;
  const int h    = bh - b * HH;
  const long base = (long)bh * TT * DD;

  // async load Q tile (64x64): 512 b128 chunks
  for (int ch = tid; ch < 512; ch += 128) {
    int row = ch >> 3, col = (ch & 7) * 8;
    async_ld_b128(&Qs[row * 64 + col], &qb[base + (long)(q0 + row) * DD + col]);
  }
  wait_async0();
  __syncthreads();

  // preload Q A-fragments for both K-steps (d = 0..31, 32..63)
  Frag16 aq[2];
#pragma unroll
  for (int ks = 0; ks < 2; ++ks) {
    const u16t* p = &Qs[(wv * 16 + lr) * 64 + ks * 32];
    aq[ks].q[0] = *(const uint4*)&p[g * 8];
    aq[ks].q[1] = *(const uint4*)&p[16 + g * 8];
  }

  float mrow[8], lrow[8];
  v8f acco[4];
#pragma unroll
  for (int r = 0; r < 8; ++r) { mrow[r] = -3.0e38f; lrow[r] = 0.f; }
#pragma unroll
  for (int nf = 0; nf < 4; ++nf) acco[nf] = zero8();

  const float scale = 0.125f;                     // 1/sqrt(64)
  const int qrow_lane = q0 + wv * 16 + 8 * g;     // + r = this lane's global q row

  for (int j0 = 0; j0 <= q0 + 63; j0 += 32) {
    // async load K chunk [32][64]
    for (int ch = tid; ch < 256; ch += 128) {
      int row = ch >> 3, col = (ch & 7) * 8;
      async_ld_b128(&Ks[row * 64 + col], &kb[base + (long)(j0 + row) * DD + col]);
    }
    // load V chunk transposed into Vt[d][kv] (register path: needs shuffle)
    for (int ch = tid; ch < 256; ch += 128) {
      int row = ch >> 3, col = (ch & 7) * 8;      // row = kv 0..31, col = d0
      uint4 t4 = *(const uint4*)&vb[base + (long)(j0 + row) * DD + col];
      const u16t* e = (const u16t*)&t4;
#pragma unroll
      for (int i = 0; i < 8; ++i) Vt[(col + i) * 40 + row] = e[i];
    }
    wait_async0();
    __syncthreads();

    if (j0 <= q0 + wv * 16 + 15) {                // wave-uniform causal skip
      // S = Q K^T : two 16x16 accums over K=64
      v8f s0 = zero8(), s1 = zero8();
#pragma unroll
      for (int ks = 0; ks < 2; ++ks) {
        Frag16 b0, b1;
        const u16t* p0 = &Ks[(lr) * 64 + ks * 32];
        b0.q[0] = *(const uint4*)&p0[g * 8];
        b0.q[1] = *(const uint4*)&p0[16 + g * 8];
        const u16t* p1 = &Ks[(16 + lr) * 64 + ks * 32];
        b1.q[0] = *(const uint4*)&p1[g * 8];
        b1.q[1] = *(const uint4*)&p1[16 + g * 8];
        s0 = __builtin_amdgcn_wmma_f32_16x16x32_bf16(false, aq[ks].v, false, b0.v, (short)0, s0, false, false);
        s1 = __builtin_amdgcn_wmma_f32_16x16x32_bf16(false, aq[ks].v, false, b1.v, (short)0, s1, false, false);
      }

      // online softmax per accumulator row (m = r + 8g)
      const int n0 = j0 + lr, n1 = j0 + 16 + lr;
#pragma unroll
      for (int r = 0; r < 8; ++r) {
        int qr = qrow_lane + r;
        float v0 = s0[r] * scale; if (n0 > qr) v0 = -3.0e38f;
        float v1 = s1[r] * scale; if (n1 > qr) v1 = -3.0e38f;
        float rmax = fmaxf(v0, v1);
#pragma unroll
        for (int off = 1; off < 16; off <<= 1) rmax = fmaxf(rmax, __shfl_xor(rmax, off, 32));
        float mnew  = fmaxf(mrow[r], rmax);
        float alpha = __expf(mrow[r] - mnew);
        float p0 = __expf(v0 - mnew);
        float p1 = __expf(v1 - mnew);
        float rsum = p0 + p1;
#pragma unroll
        for (int off = 1; off < 16; off <<= 1) rsum += __shfl_xor(rsum, off, 32);
        lrow[r] = lrow[r] * alpha + rsum;
        mrow[r] = mnew;
#pragma unroll
        for (int nf = 0; nf < 4; ++nf) acco[nf][r] *= alpha;
        int prow = (wv * 16 + r + 8 * g) * 40;    // stage P in A-frag source layout
        Ps[prow + lr]      = f2bf(p0);
        Ps[prow + 16 + lr] = f2bf(p1);
      }

      // O += P V  (A = P 16x32 from LDS, B = Vt rows = output dims)
      Frag16 pa;
      const u16t* pp = &Ps[(wv * 16 + lr) * 40];
      pa.q[0] = *(const uint4*)&pp[g * 8];
      pa.q[1] = *(const uint4*)&pp[16 + g * 8];
#pragma unroll
      for (int nf = 0; nf < 4; ++nf) {
        Frag16 vf;
        const u16t* vp = &Vt[(nf * 16 + lr) * 40];
        vf.q[0] = *(const uint4*)&vp[g * 8];
        vf.q[1] = *(const uint4*)&vp[16 + g * 8];
        acco[nf] = __builtin_amdgcn_wmma_f32_16x16x32_bf16(
            false, pa.v, false, vf.v, (short)0, acco[nf], false, false);
      }
    }
    __syncthreads();
  }

  // normalize and store into (B,T,C) bf16 buffer for the output projection
#pragma unroll
  for (int r = 0; r < 8; ++r) {
    float inv = 1.0f / lrow[r];
    int t = q0 + wv * 16 + r + 8 * g;
#pragma unroll
    for (int nf = 0; nf < 4; ++nf) {
      int d = nf * 16 + lr;
      aout[(long)(b * TT + t) * CC + h * DD + d] = f2bf(acco[nf][r] * inv);
    }
  }
}

// ---------------------------------------------------------------------------
// Kernel: output projection  out[m,n] = sum_k aout[m,k] * wproj[n,k]  (fp32 out)
// Same double-buffered async pipeline as gemm_qkv.
// ---------------------------------------------------------------------------
__global__ void __launch_bounds__(256) gemm_proj(const u16t* __restrict__ A,
                                                 const u16t* __restrict__ W,
                                                 float* __restrict__ out) {
  const int K = CC;
  __shared__ alignas(16) u16t At[2][128 * 32];
  __shared__ alignas(16) u16t Bt[2][128 * 32];
  const int tid  = threadIdx.x;
  const int lane = tid & 31;
  const int wv   = tid >> 5;
  const int mw   = wv >> 2;
  const int nw   = wv & 3;
  const int g    = lane >> 4;
  const int lr   = lane & 15;
  const int bm   = blockIdx.y * 128;
  const int bn   = blockIdx.x * 128;

  v8f acc[4][2];
#pragma unroll
  for (int i = 0; i < 4; ++i)
#pragma unroll
    for (int j = 0; j < 2; ++j) acc[i][j] = zero8();

  auto issue_tile = [&](int kt, int buf) {
#pragma unroll
    for (int c = 0; c < 2; ++c) {
      int ch  = tid + c * 256;
      int row = ch >> 2;
      int col = (ch & 3) * 8;
      async_ld_b128(&At[buf][row * 32 + col], &A[(long)(bm + row) * K + kt + col]);
      async_ld_b128(&Bt[buf][row * 32 + col], &W[(long)(bn + row) * K + kt + col]);
    }
  };

  issue_tile(0, 0);
  int cur = 0;
  for (int kt = 0; kt < K; kt += 32) {
    wait_async0();
    __syncthreads();
    if (kt + 32 < K) issue_tile(kt + 32, cur ^ 1);
    if (kt + 64 < K) {
      __builtin_prefetch(&A[(long)(bm + (tid >> 1)) * K + kt + 64], 0, 1);
      __builtin_prefetch(&W[(long)(bn + (tid >> 1)) * K + kt + 64], 0, 1);
    }

    const u16t* Ab = At[cur];
    const u16t* Bb = Bt[cur];
    Frag16 af[4], bfr[2];
#pragma unroll
    for (int mf = 0; mf < 4; ++mf) {
      const u16t* p = &Ab[(mw * 64 + mf * 16 + lr) * 32];
      af[mf].q[0] = *(const uint4*)&p[g * 8];
      af[mf].q[1] = *(const uint4*)&p[16 + g * 8];
    }
#pragma unroll
    for (int nf = 0; nf < 2; ++nf) {
      const u16t* p = &Bb[(nw * 32 + nf * 16 + lr) * 32];
      bfr[nf].q[0] = *(const uint4*)&p[g * 8];
      bfr[nf].q[1] = *(const uint4*)&p[16 + g * 8];
    }
#pragma unroll
    for (int mf = 0; mf < 4; ++mf)
#pragma unroll
      for (int nf = 0; nf < 2; ++nf)
        acc[mf][nf] = __builtin_amdgcn_wmma_f32_16x16x32_bf16(
            false, af[mf].v, false, bfr[nf].v, (short)0, acc[mf][nf], false, false);
    cur ^= 1;
  }

#pragma unroll
  for (int mf = 0; mf < 4; ++mf)
#pragma unroll
    for (int nf = 0; nf < 2; ++nf)
#pragma unroll
      for (int r = 0; r < 8; ++r) {
        int gm = bm + mw * 64 + mf * 16 + r + 8 * g;
        int gn = bn + nw * 32 + nf * 16 + lr;
        out[(long)gm * CC + gn] = acc[mf][nf][r];
      }
}

// ---------------------------------------------------------------------------
// Launch
// ---------------------------------------------------------------------------
extern "C" void kernel_launch(void* const* d_in, const int* in_sizes, int n_in,
                              void* d_out, int out_size, void* d_ws, size_t ws_size,
                              hipStream_t stream) {
  const float* x      = (const float*)d_in[0];   // (B,T,C)
  const float* w_qkv  = (const float*)d_in[1];   // (3C,C)
  const float* w_proj = (const float*)d_in[2];   // (C,C)
  const float* rsin   = (const float*)d_in[3];   // (T,D/2)
  const float* rcos   = (const float*)d_in[4];   // (T,D/2)
  float* out = (float*)d_out;

  char* ws = (char*)d_ws;
  u16t* xbf    = (u16t*)(ws);                    // 8192*768
  u16t* wqkvbf = (u16t*)(ws + 12582912);         // 2304*768
  u16t* wprjbf = (u16t*)(ws + 16121856);         // 768*768
  u16t* qb     = (u16t*)(ws + 17301504);         // (B,H,T,D)
  u16t* kb     = (u16t*)(ws + 29884416);
  u16t* vb     = (u16t*)(ws + 42467328);
  u16t* aout   = (u16t*)(ws + 55050240);         // (B,T,C)

  // 1) casts to bf16
  {
    int n = BT * CC;
    cast_f32_bf16<<<(n + 255) / 256, 256, 0, stream>>>(x, xbf, n);
    n = 3 * CC * CC;
    cast_f32_bf16<<<(n + 255) / 256, 256, 0, stream>>>(w_qkv, wqkvbf, n);
    n = CC * CC;
    cast_f32_bf16<<<(n + 255) / 256, 256, 0, stream>>>(w_proj, wprjbf, n);
  }
  // 2) QKV projection (WMMA, async double-buffered) with scatter to (B,H,T,D)
  gemm_qkv<<<dim3(18, 64), 256, 0, stream>>>(xbf, wqkvbf, qb, kb, vb);
  // 3) RoPE on q, k
  {
    int n = 4 * HH * TT * 32;
    rope_kernel<<<(n + 255) / 256, 256, 0, stream>>>(qb, kb, rsin, rcos);
  }
  // 4) causal flash attention (WMMA + async K/Q loads)
  flash_attn<<<dim3(TT / 64, 4 * HH), 128, 0, stream>>>(qb, kb, vb, aout);
  // 5) output projection (WMMA, async double-buffered) -> fp32
  gemm_proj<<<dim3(6, 64), 256, 0, stream>>>(aout, wprjbf, out);
}
